// AttentionQ_24893630448192
// MI455X (gfx1250) — matrix operand order repaired and checked by reading, not compiled, not run on hardware
//
#include <hip/hip_runtime.h>

typedef __attribute__((ext_vector_type(16))) __bf16       v16bf;
typedef __attribute__((ext_vector_type(8)))  float        v8f;
typedef __attribute__((ext_vector_type(4)))  float        v4f;
typedef __attribute__((ext_vector_type(8)))  unsigned int v8u;
typedef __attribute__((ext_vector_type(4)))  unsigned int v4u;
typedef __attribute__((ext_vector_type(2)))  unsigned int v2u;

namespace {
constexpr int kBatch   = 64;
constexpr int kN       = 2048;
constexpr int kD       = 256;
constexpr int kF       = 128;   // NUM_INDS
constexpr int kBins    = 32;
constexpr int kRowsWG  = 128;   // 8 waves x 16 rows
constexpr int kChunks  = kN / kRowsWG;   // 16
constexpr int kKPanel  = 128;   // K columns of I staged per LDS panel
constexpr int kPad     = 136;   // padded bf16 row stride in LDS (bank spread)
constexpr int kOutPerB = kF * kBins;     // 4096
}

// Pack two f32 into one dword of bf16x2 (round-bias + one v_perm_b32).
__device__ __forceinline__ unsigned int pack_bf16x2(float lo, float hi) {
    unsigned int ulo = __float_as_uint(lo) + 0x8000u;
    unsigned int uhi = __float_as_uint(hi) + 0x8000u;
    // D = {uhi[3], uhi[2], ulo[3], ulo[2]}
    return __builtin_amdgcn_perm(uhi, ulo, 0x07060302u);
}

__global__ void zero_f32_kernel(float* __restrict__ p, int n) {
    int i = blockIdx.x * blockDim.x + threadIdx.x;
    if (i < n) p[i] = 0.0f;
}

__global__ __launch_bounds__(256)
void sigmoid_hist_wmma_kernel(const float* __restrict__ X,
                              const float* __restrict__ I,
                              float* __restrict__ out) {
    __shared__ __align__(16) unsigned short sI[kF * kPad];  // bf16 panel of I
    __shared__ unsigned int hist[kF * kBins];

    const int tid   = threadIdx.x;
    const int wg    = blockIdx.x;
    const int b     = wg / kChunks;
    const int chunk = wg % kChunks;
    const int wave  = tid >> 5;
    const int lane  = tid & 31;
    const int hlane = lane & 15;           // column / row-within-halfwave index
    const int kgrp8  = (lane >> 4) * 8;    // A-fragment K group offset
    const int kgrp16 = (lane >> 4) * 16;   // unused; B fragments also use kgrp8

    // zero per-WG histogram
    for (int t = tid; t < kF * kBins; t += 256) hist[t] = 0u;

    const int row   = chunk * kRowsWG + wave * 16 + hlane;  // A-fragment row
    const float* xrow = X + ((size_t)b * kN + row) * kD;

    // --- software-pipelined A loads: issue step 0 before staging I ---
    v4f c0 = *(const v4f*)(xrow + kgrp8);
    v4f c1 = *(const v4f*)(xrow + kgrp8 + 4);
    v4f c2 = *(const v4f*)(xrow + kgrp8 + 16);
    v4f c3 = *(const v4f*)(xrow + kgrp8 + 20);

    v8f zero = {};
    v8f acc[8];
#pragma unroll
    for (int nt = 0; nt < 8; ++nt) acc[nt] = zero;

#pragma unroll
    for (int khalf = 0; khalf < 2; ++khalf) {
        __syncthreads();  // protect sI reuse (also covers hist zeroing once)
        // stage I[:, khalf*128 .. +128) into LDS as bf16x2 pairs (b64 stores)
        for (int t = tid; t < (kF * kKPanel) / 4; t += 256) {
            const int e = t * 4;
            const int f = e >> 7;          // /128
            const int d = e & 127;
            v4f v = *(const v4f*)(I + f * kD + khalf * kKPanel + d);
            v2u p = { pack_bf16x2(v[0], v[1]), pack_bf16x2(v[2], v[3]) };
            *(v2u*)(sI + f * kPad + d) = p;   // ds_store_b64
        }
        __syncthreads();

#pragma unroll
        for (int s = 0; s < 4; ++s) {
            // Issue NEXT k-step's X loads before consuming the current block,
            // so global latency overlaps the pack + 8 WMMAs below.
            v4f n0, n1, n2, n3;
            const bool has_next = (khalf == 0) || (s < 3);
            const int knext = khalf * kKPanel + s * 32 + 32 + kgrp8;
            if (has_next) {
                n0 = *(const v4f*)(xrow + knext);
                n1 = *(const v4f*)(xrow + knext + 4);
                n2 = *(const v4f*)(xrow + knext + 16);
                n3 = *(const v4f*)(xrow + knext + 20);
                if (knext + 32 < kD)
                    __builtin_prefetch(xrow + knext + 32, 0, 1);  // 2 steps ahead
            }

            // A fragment: row = hlane, K = kglob..+7, kglob+16..+23 (bf16)
            v8u au;
            au[0] = pack_bf16x2(c0[0], c0[1]);
            au[1] = pack_bf16x2(c0[2], c0[3]);
            au[2] = pack_bf16x2(c1[0], c1[1]);
            au[3] = pack_bf16x2(c1[2], c1[3]);
            au[4] = pack_bf16x2(c2[0], c2[1]);
            au[5] = pack_bf16x2(c2[2], c2[3]);
            au[6] = pack_bf16x2(c3[0], c3[1]);
            au[7] = pack_bf16x2(c3[2], c3[3]);
            v16bf afrag = __builtin_bit_cast(v16bf, au);

            // B fragments: column = hlane, K = s*32 + kgrp8 .. +7, +16 .. +23
            const int kB16 = s * 32 + kgrp8;
#pragma unroll
            for (int nt = 0; nt < 8; ++nt) {
                const unsigned short* bp = sI + (nt * 16 + hlane) * kPad + kB16;
                v4u blo = *(const v4u*)bp;         // ds_load_b128
                v4u bhi = *(const v4u*)(bp + 16);  // ds_load_b128
                v8u bu;
                bu.lo = blo;
                bu.hi = bhi;
                v16bf bfrag = __builtin_bit_cast(v16bf, bu);
                acc[nt] = __builtin_amdgcn_wmma_f32_16x16x32_bf16(
                    false, afrag, false, bfrag, (short)0, acc[nt], false, false);
            }

            if (has_next) { c0 = n0; c1 = n1; c2 = n2; c3 = n3; }
        }
    }

    // sigmoid + hard binning into LDS counters
#pragma unroll
    for (int nt = 0; nt < 8; ++nt) {
        const int f = nt * 16 + hlane;  // D layout: N = lane%16
#pragma unroll
        for (int r = 0; r < 8; ++r) {
            float sc = acc[nt][r];
            float pr = 1.0f / (1.0f + __expf(-sc));
            int bin = (int)(pr * 32.0f);
            bin = bin < 0 ? 0 : (bin > kBins - 1 ? kBins - 1 : bin);
            atomicAdd(&hist[f * kBins + bin], 1u);  // ds_add_u32
        }
    }
    __syncthreads();

    // flush normalized counts to global (chunks accumulate via f32 atomics)
    const float scale = 1.0f / (float)kN;
    float* dst = out + (size_t)b * kOutPerB;
    for (int t = tid; t < kF * kBins; t += 256) {
        unsigned int c = hist[t];
        if (c) atomicAdd(dst + t, (float)c * scale);  // global_atomic_add_f32
    }
}

extern "C" void kernel_launch(void* const* d_in, const int* in_sizes, int n_in,
                              void* d_out, int out_size, void* d_ws, size_t ws_size,
                              hipStream_t stream) {
    (void)in_sizes; (void)n_in; (void)d_ws; (void)ws_size;
    const float* X = (const float*)d_in[0];   // (64, 2048, 256) f32
    const float* I = (const float*)d_in[1];   // (1, 128, 256)  f32
    float* out = (float*)d_out;               // (64, 4096)     f32

    zero_f32_kernel<<<(out_size + 255) / 256, 256, 0, stream>>>(out, out_size);
    sigmoid_hist_wmma_kernel<<<kBatch * kChunks, 256, 0, stream>>>(X, I, out);
}
